// VectorQuantizer_39367670235391
// MI455X (gfx1250) — matrix operand order repaired and checked by reading, not compiled, not run on hardware
//
#include <hip/hip_runtime.h>
#include <hip/hip_bf16.h>

// Problem dims (fixed by reference): B=32, S=1024, D=512, K=8192
#define DD 512
#define KK 8192
#define NROWS 32768  // B*S

typedef __attribute__((ext_vector_type(16))) __bf16 v16bf;
typedef __attribute__((ext_vector_type(8)))  float  v8f;

// ---------------------------------------------------------------------------
// Kernel 0: zero the code-usage histogram.
// ---------------------------------------------------------------------------
__global__ __launch_bounds__(256) void vq_init_kernel(int* __restrict__ counts) {
    int i = blockIdx.x * 256 + threadIdx.x;
    if (i < KK) counts[i] = 0;
}

// ---------------------------------------------------------------------------
// Kernel 1: per-code squared norms ||e_k||^2 (one wave per codebook row,
// float4 vectorized: 4 iterations of global_load_b128 per lane).
// ---------------------------------------------------------------------------
__global__ __launch_bounds__(256) void vq_enorm_kernel(const float* __restrict__ emb,
                                                       float* __restrict__ enorm) {
    const int lane = threadIdx.x & 31;
    const int wave = threadIdx.x >> 5;
    const int row  = blockIdx.x * 8 + wave;
    const float4* p = (const float4*)(emb + (size_t)row * DD);
    float s = 0.f;
    #pragma unroll
    for (int c = 0; c < DD / 4; c += 32) {
        const float4 v = p[c + lane];
        s += v.x * v.x + v.y * v.y + v.z * v.z + v.w * v.w;
    }
    #pragma unroll
    for (int off = 16; off >= 1; off >>= 1) s += __shfl_xor(s, off, 32);
    if (lane == 0) enorm[row] = s;
}

// ---------------------------------------------------------------------------
// Kernel 1b: pack codebook into WMMA-native bf16 B fragments, done ONCE.
// Layout: embPack[jt][kk][l][s] = bf16(emb[jt*16+l%16][kk*32+k]), k = 8*(l/16)+(s&7)+16*(s>>3)
//   (jt = column tile 0..511, kk = k-step 0..15, l = lane 0..31,
//    s = fragment element 0..15). A whole 32x16 B fragment is 1 KB
//    contiguous; each lane's 16 elements are a 32-byte contiguous chunk.
// ---------------------------------------------------------------------------
__global__ __launch_bounds__(256) void vq_pack_kernel(const float* __restrict__ emb,
                                                      __bf16* __restrict__ embPack) {
    __shared__ __bf16 tile[16][DD + 8];
    const int jt  = blockIdx.x;
    const int tid = threadIdx.x;
    for (int i = tid; i < 16 * DD; i += 256) {
        const int n = i >> 9;          // /512
        const int d = i & (DD - 1);
        tile[n][d] = (__bf16)emb[(size_t)(jt * 16 + n) * DD + d];
    }
    __syncthreads();
    __bf16* outp = embPack + (size_t)jt * (16 * 32 * 16);
    for (int i = tid; i < 16 * 32 * 16; i += 256) {   // coalesced writes
        const int s  = i & 15;
        const int l  = (i >> 4) & 31;
        const int kk = i >> 9;
        const int n  = l & 15;
        const int k  = ((l >> 4) << 3) + (s & 7) + ((s >> 3) << 4);
        outp[i] = tile[n][kk * 32 + k];
    }
}

// ---------------------------------------------------------------------------
// Kernel 2: fused bf16-WMMA distance GEMM + running argmin.
// One workgroup (128 thr = 4 waves) owns a 16-row tile of X, staged bf16 in
// LDS, then hoisted into 16 register-resident A fragments (128 VGPRs).
// Steady state per k-step: 2x global_load_b128 (packed B) + 1x v_wmma.
// Each wave streams an interleaved quarter of the 512 column tiles, folding
// score = ||e||^2 - 2*(x.e) into a per-lane running (min, argmin).
// ---------------------------------------------------------------------------
__global__ __launch_bounds__(128) void vq_argmin_kernel(const float* __restrict__ x,
                                                        const __bf16* __restrict__ embPack,
                                                        const float* __restrict__ enorm,
                                                        int* __restrict__ outIdx,
                                                        int* __restrict__ counts) {
    __shared__ __bf16 Asl[16][DD + 8];   // +8 bf16 pad -> LDS bank spread
    __shared__ float  wmin[4][16];
    __shared__ int    widx[4][16];

    const int tid  = threadIdx.x;
    const int lane = tid & 31;
    const int wave = tid >> 5;
    const int row0 = blockIdx.x * 16;

    // Stage the 16x512 f32 row tile into LDS as bf16 (coalesced, converted).
    for (int i = tid; i < 16 * DD; i += 128) {
        const int r = i >> 9;          // /512
        const int c = i & (DD - 1);
        Asl[r][c] = (__bf16)x[(size_t)(row0 + r) * DD + c];
    }
    __syncthreads();

    // ISA fragment mappings (CDNA5 ISA):
    //   A 16x32 bf16 : lane -> M = lane&15 ; K-offset 0 (lanes 0-15) / 8 (16-31)
    //   B 32x16 bf16 : lane -> N = lane&15 ; K-offset 0 (lanes 0-15) / 8 (16-31)
    //   C 16x16 f32  : lane -> N = lane&15 ; VGPR r -> M = r + 8*(lane>=16)
    const int m  = lane & 15;
    const int ko = (lane < 16) ? 0 : 8;

    // Hoist all 16 A fragments into registers: invariant over column tiles.
    v16bf afrag[16];
    #pragma unroll
    for (int kk = 0; kk < DD / 32; ++kk) {
        const int kb = kk * 32;
        #pragma unroll
        for (int e = 0; e < 8; ++e) {
            afrag[kk][e]     = Asl[m][kb + ko + e];
            afrag[kk][e + 8] = Asl[m][kb + 16 + ko + e];
        }
    }

    float bestv[8];
    int   besti[8];
    #pragma unroll
    for (int r = 0; r < 8; ++r) { bestv[r] = 3.402823e38f; besti[r] = 0; }

    for (int jt = wave; jt < KK / 16; jt += 4) {
        // This lane's 32-byte B chunk within each packed 1 KB fragment.
        const __bf16* bp = embPack + (size_t)jt * (16 * 32 * 16) + lane * 16;
        v8f acc = {};
        #pragma unroll
        for (int kk = 0; kk < DD / 32; ++kk) {
            const v16bf b = *(const v16bf*)(bp + kk * (32 * 16));
            acc = __builtin_amdgcn_wmma_f32_16x16x32_bf16(
                false, afrag[kk], false, b, (short)0, acc, false, false);
        }
        const int   col0  = jt * 16;
        const int   myCol = col0 + (lane & 15);
        const float en    = enorm[myCol];
        #pragma unroll
        for (int r = 0; r < 8; ++r) {
            const float s = en - 2.0f * acc[r];
            if (s < bestv[r] || (s == bestv[r] && myCol < besti[r])) {
                bestv[r] = s; besti[r] = myCol;
            }
        }
    }

    // Argmin across the 16 lanes of each half (columns live across lanes).
    #pragma unroll
    for (int r = 0; r < 8; ++r) {
        float bv = bestv[r]; int bi = besti[r];
        #pragma unroll
        for (int off = 8; off >= 1; off >>= 1) {   // stays within 16-lane half
            const float ov = __shfl_xor(bv, off, 32);
            const int   oi = __shfl_xor(bi, off, 32);
            if (ov < bv || (ov == bv && oi < bi)) { bv = ov; bi = oi; }
        }
        bestv[r] = bv; besti[r] = bi;
    }
    if ((lane & 15) == 0) {
        const int half = lane >> 4;                 // half -> rows r + 8*half
        #pragma unroll
        for (int r = 0; r < 8; ++r) {
            wmin[wave][half * 8 + r] = bestv[r];
            widx[wave][half * 8 + r] = besti[r];
        }
    }
    __syncthreads();

    // Combine the 4 waves (disjoint column sets) and commit.
    if (tid < 16) {
        float bv = wmin[0][tid]; int bi = widx[0][tid];
        #pragma unroll
        for (int w = 1; w < 4; ++w) {
            const float ov = wmin[w][tid]; const int oi = widx[w][tid];
            if (ov < bv || (ov == bv && oi < bi)) { bv = ov; bi = oi; }
        }
        outIdx[row0 + tid] = bi;
        atomicAdd(&counts[bi], 1);
    }
}

// ---------------------------------------------------------------------------
// Kernel 3: gather quantized rows (f32, full precision), emit quantized_st
// (forward value == quantized), per-row sum of (q-x)^2, index as float.
// float4-vectorized: one b128 load/store pair per thread.
// ---------------------------------------------------------------------------
__global__ __launch_bounds__(128) void vq_gather_kernel(const float* __restrict__ x,
                                                        const float* __restrict__ emb,
                                                        const int* __restrict__ idx,
                                                        float* __restrict__ outQ,
                                                        float* __restrict__ outIdxF,
                                                        float* __restrict__ rowLoss) {
    __shared__ float part[4];
    const int row  = blockIdx.x;
    const int tid  = threadIdx.x;
    const int code = idx[row];
    const float4* xr = (const float4*)(x   + (size_t)row  * DD);
    const float4* er = (const float4*)(emb + (size_t)code * DD);
    float4*       qr = (float4*)(outQ + (size_t)row * DD);
    const float4 q = er[tid];
    const float4 v = xr[tid];
    qr[tid] = q;
    const float dx = q.x - v.x, dy = q.y - v.y, dz = q.z - v.z, dw = q.w - v.w;
    float s = dx * dx + dy * dy + dz * dz + dw * dw;
    #pragma unroll
    for (int off = 16; off >= 1; off >>= 1) s += __shfl_xor(s, off, 32);
    if ((tid & 31) == 0) part[tid >> 5] = s;
    __syncthreads();
    if (tid == 0) {
        rowLoss[row] = part[0] + part[1] + part[2] + part[3];
        outIdxF[row] = (float)code;
    }
}

// ---------------------------------------------------------------------------
// Kernel 4: deterministic fixed-order reduction -> loss & perplexity scalars.
// loss = 1.25 * mean((q-x)^2);  perplexity = exp(-sum p*log(p+1e-10)).
// ---------------------------------------------------------------------------
__global__ __launch_bounds__(256) void vq_finalize_kernel(const int* __restrict__ counts,
                                                          const float* __restrict__ rowLoss,
                                                          float* __restrict__ outLoss,
                                                          float* __restrict__ outPerp) {
    __shared__ float shL[256];
    __shared__ float shH[256];
    const int tid = threadIdx.x;
    float l = 0.f;
    for (int i = tid; i < NROWS; i += 256) l += rowLoss[i];
    float h = 0.f;
    const float invN = 1.0f / (float)NROWS;
    for (int k = tid; k < KK; k += 256) {
        const float p = (float)counts[k] * invN;
        h += p * logf(p + 1e-10f);
    }
    shL[tid] = l; shH[tid] = h;
    __syncthreads();
    for (int s = 128; s >= 1; s >>= 1) {
        if (tid < s) { shL[tid] += shL[tid + s]; shH[tid] += shH[tid + s]; }
        __syncthreads();
    }
    if (tid == 0) {
        *outLoss = 1.25f * shL[0] / (float)((size_t)NROWS * DD);
        *outPerp = expf(-shH[0]);
    }
}

// ---------------------------------------------------------------------------
extern "C" void kernel_launch(void* const* d_in, const int* in_sizes, int n_in,
                              void* d_out, int out_size, void* d_ws, size_t ws_size,
                              hipStream_t stream) {
    const float* x   = (const float*)d_in[0];   // [N, D] f32
    const float* emb = (const float*)d_in[1];   // [K, D] f32

    // Output layout (tuple flattened): quantized_st | loss | perplexity | indices
    float* outQ    = (float*)d_out;
    float* outLoss = outQ + (size_t)NROWS * DD;          // 16777216
    float* outPerp = outLoss + 1;                        // 16777217
    float* outIdxF = outPerp + 1;                        // 16777218 .. +N

    // Workspace carve-up (~8.7 MB total)
    char*   ws      = (char*)d_ws;
    float*  enorm   = (float*)(ws);                                   // K f32
    int*    counts  = (int*)  (ws + (size_t)KK * 4);                  // K i32
    int*    idx     = (int*)  (ws + (size_t)KK * 8);                  // N i32
    float*  rowLoss = (float*)(ws + (size_t)KK * 8 + (size_t)NROWS * 4);  // N f32
    __bf16* embPack = (__bf16*)(ws + (size_t)KK * 8 + (size_t)NROWS * 8); // K*D bf16

    vq_init_kernel<<<(KK + 255) / 256, 256, 0, stream>>>(counts);
    vq_enorm_kernel<<<KK / 8, 256, 0, stream>>>(emb, enorm);
    vq_pack_kernel<<<KK / 16, 256, 0, stream>>>(emb, embPack);
    vq_argmin_kernel<<<NROWS / 16, 128, 0, stream>>>(x, embPack, enorm, idx, counts);
    vq_gather_kernel<<<NROWS, 128, 0, stream>>>(x, emb, idx, outQ, outIdxF, rowLoss);
    vq_finalize_kernel<<<1, 256, 0, stream>>>(counts, rowLoss, outLoss, outPerp);
}
